// TestCUDAMoe_81604378624117
// MI455X (gfx1250) — matrix operand order repaired and checked by reading, not compiled, not run on hardware
//
#include <hip/hip_runtime.h>

// ---------------- problem constants ----------------
#define NE      32
#define EW      2944
#define EW_EFF  2880      // ORIG: columns >= 2880 are zeroed by _act
#define TOPK    4
#define T_D     8
#define MAXNNZ  256
#define HID     2048
#define NTOK    1024
#define ALPHA   1.702f
#define LIMIT   7.0f
#define BK      64        // K-chunk staged in LDS

typedef _Float16 f16;
typedef __attribute__((ext_vector_type(8)))  _Float16 v8h;
typedef __attribute__((ext_vector_type(16))) _Float16 v16h;
typedef __attribute__((ext_vector_type(8)))  float    v8f;

// ---------------- workspace layout (bytes) ----------------
#define OFF_RW      0u                          // 1024*32 f32      = 131072
#define OFF_CDCS    131072u                     // 16 int
#define OFF_MASKR   131328u                     // 256*8 int        = 8192
#define OFF_OUTACC  139520u                     // 1024*2048 f32    = 8388608
#define OFF_ACTD    8528128u                    // 1024*8*2880 f16  = 47185920
#define OFF_ACTS    55714048u                   // 256*8*2880 f16   = 11796480

// ---------------- CDNA5 async copy + WMMA helpers ----------------
// GLOBAL_LOAD_ASYNC_TO_LDS_B128, GVS mode: SGPR 64-bit base + per-lane 32-bit byte offset.
__device__ __forceinline__ void async_ld_b128(unsigned lds_off, unsigned goff,
                                              unsigned long long sbase) {
    asm volatile("global_load_async_to_lds_b128 %0, %1, %2"
                 :: "v"(lds_off), "v"(goff), "s"(sbase) : "memory");
}
__device__ __forceinline__ void wait_async_le2() { asm volatile("s_wait_asynccnt 0x2" ::: "memory"); }
__device__ __forceinline__ void wait_async_0()  { asm volatile("s_wait_asynccnt 0x0" ::: "memory"); }

// A (16x32 f16) fragment: lane holds row m=lane&15; K contiguous at 2*kbA .. 2*kbA+15.
__device__ __forceinline__ v16h a_frag(const f16* __restrict__ row, int k, int kbA) {
    v8h lo = *(const v8h*)(row + k + kbA * 2);
    v8h hi = *(const v8h*)(row + k + kbA * 2 + 8);
    return __builtin_shufflevector(lo, hi, 0,1,2,3,4,5,6,7,8,9,10,11,12,13,14,15);
}
__device__ __forceinline__ v8f wmma_f16(v16h a, v16h b, v8f c) {
    return __builtin_amdgcn_wmma_f32_16x16x32_f16(false, a, false, b, (short)0, c, false, false);
}
__device__ __forceinline__ float glu_act(float up, float gate) {
    float g = fminf(gate, LIMIT);
    float u = fminf(fmaxf(up, -LIMIT), LIMIT);
    float glu = g / (1.0f + __expf(-ALPHA * g));
    return (u + 1.0f) * glu;
}

// ---------------- 1. router ----------------
__global__ void k_router(const f16* __restrict__ x, const f16* __restrict__ gw,
                         float* __restrict__ rw) {
    const int n = blockIdx.x;
    const int e = threadIdx.x;           // 0..31, one wave32
    const f16* xr = x + (long)n * HID;
    const f16* wr = gw + (long)e * HID;
    float acc = 0.f;
    for (int k = 0; k < HID; k += 8) {
        v8h xv = *(const v8h*)(xr + k);
        v8h wv = *(const v8h*)(wr + k);
        #pragma unroll
        for (int j = 0; j < 8; ++j) acc += (float)xv[j] * (float)wv[j];
    }
    float m = acc;
    #pragma unroll
    for (int off = 16; off; off >>= 1) m = fmaxf(m, __shfl_xor(m, off, 32));
    float p = __expf(acc - m);
    float s = p;
    #pragma unroll
    for (int off = 16; off; off >>= 1) s += __shfl_xor(s, off, 32);
    float prob = p / s;

    bool chosen = false;
    float total = 0.f;
    #pragma unroll
    for (int j = 0; j < TOPK; ++j) {
        float v = chosen ? -1.0f : prob;
        int   bi = e;
        #pragma unroll
        for (int off = 16; off; off >>= 1) {
            float ov = __shfl_xor(v,  off, 32);
            int   oi = __shfl_xor(bi, off, 32);
            if (ov > v || (ov == v && oi < bi)) { v = ov; bi = oi; }
        }
        if (e == bi) chosen = true;
        total += v;
    }
    rw[(long)n * NE + e] = chosen ? (prob / total) : 0.0f;
}

// ---------------- 2. expert ranking ----------------
__global__ void k_expert_rank(const float* __restrict__ rw, int* __restrict__ cdcs) {
    __shared__ int cnt[NE];
    const int e = threadIdx.x;
    int c = 0;
    for (int n = 0; n < NTOK; ++n) c += (rw[(long)n * NE + e] != 0.f) ? 1 : 0;
    cnt[e] = c;
    __syncthreads();
    int rank = 0;
    for (int j = 0; j < NE; ++j) {
        int cj = cnt[j];
        if (cj > c || (cj == c && j < e)) ++rank;
    }
    if (rank < 2 * T_D) cdcs[rank] = e;
}

// ---------------- 3. top-256 tokens per sparse expert ----------------
__global__ void k_maskr(const float* __restrict__ rw, const int* __restrict__ cdcs,
                        int* __restrict__ maskr) {
    __shared__ float vals[NTOK];
    const int t = blockIdx.x;
    const int n = threadIdx.x;
    const int e = cdcs[T_D + t];
    float v = rw[(long)n * NE + e];
    vals[n] = v;
    __syncthreads();
    int rank = 0;
    for (int j = 0; j < NTOK; ++j) {
        float vj = vals[j];
        if (vj > v || (vj == v && j < n)) ++rank;
    }
    if (rank < MAXNNZ) maskr[rank * T_D + t] = n;
}

// ---------------- 4. fused up+gate GEMM + activation ----------------
// Block 256 thr: 64 tokens x 64 w (U and G). 8 waves = 2 M-groups x 4 N-tiles.
// A tile (64 x 64 f16) double-buffered in LDS via async copies; B streamed from global.
// Chunk body: ALL fragment loads first, then a straight run of 8 in-place WMMAs.
__global__ void __launch_bounds__(256, 1)
k_upgate(const f16* __restrict__ x, const f16* __restrict__ u,
         const f16* __restrict__ g, const f16* __restrict__ ub,
         const f16* __restrict__ gb, const float* __restrict__ rw,
         const int* __restrict__ cdcs, const int* __restrict__ maskr,
         f16* __restrict__ act, int sparse) {
    __shared__ __align__(16) f16 atile[2][64][BK];
    const int tid = threadIdx.x;
    const int wave = tid >> 5, lane = tid & 31;
    const int mlane = lane & 15;
    const int kbA = (lane >> 4) << 3;
    const int kbB = (lane >> 4) << 4;
    const int t = blockIdx.z;
    const int e = cdcs[sparse ? (T_D + t) : t];

    // staging: each thread owns rows srow, srow+32 at column scol (8 f16 = 16B)
    const int srow = tid >> 3;
    const int scol = (tid & 7) * 8;
    const int gr0 = blockIdx.y * 64 + srow;
    const int gr1 = gr0 + 32;
    const unsigned goff0 = ((sparse ? maskr[gr0 * T_D + t] : gr0) * HID + scol) * 2u;
    const unsigned goff1 = ((sparse ? maskr[gr1 * T_D + t] : gr1) * HID + scol) * 2u;
    const unsigned long long xbase = (unsigned long long)x;
    const unsigned lds0 = (unsigned)(unsigned long long)&atile[0][srow][scol];

    // compute: wave (mi,ni); two 16-row M tiles, one 16-col N tile, U and G
    const int ni = wave & 3, mi = wave >> 2;
    const int w0 = blockIdx.x * 64 + ni * 16 + mlane;
    const f16* burow = u + ((long)e * EW + w0) * HID;
    const f16* bgrow = g + ((long)e * EW + w0) * HID;
    const int lr0 = mi * 32 + mlane, lr1 = lr0 + 16;

    v8f aU0 = {0,0,0,0,0,0,0,0}, aG0 = aU0, aU1 = aU0, aG1 = aU0;

    async_ld_b128(lds0,        goff0, xbase);
    async_ld_b128(lds0 + 4096, goff1, xbase);

    const int NCH = HID / BK;   // 32
    for (int c = 0; c < NCH; ++c) {
        if (c + 1 < NCH) {
            const unsigned lb = lds0 + (((c + 1) & 1) ? 8192u : 0u);
            const unsigned kB = (unsigned)((c + 1) * BK * 2);
            async_ld_b128(lb,        goff0 + kB, xbase);
            async_ld_b128(lb + 4096, goff1 + kB, xbase);
            wait_async_le2();
        } else {
            wait_async_0();
        }
        __syncthreads();
        const int kc = c * BK;
        const f16* ap0 = &atile[c & 1][lr0][0];
        const f16* ap1 = &atile[c & 1][lr1][0];
        // -------- load all fragments for this chunk --------
        v16h a00 = a_frag(ap0, 0,  kbA);
        v16h a10 = a_frag(ap1, 0,  kbA);
        v16h a01 = a_frag(ap0, 32, kbA);
        v16h a11 = a_frag(ap1, 32, kbA);
        v16h bu0 = *(const v16h*)(burow + kc + 0  + kbB);
        v16h bg0 = *(const v16h*)(bgrow + kc + 0  + kbB);
        v16h bu1 = *(const v16h*)(burow + kc + 32 + kbB);
        v16h bg1 = *(const v16h*)(bgrow + kc + 32 + kbB);
        // -------- straight WMMA run, round-robin over accumulators --------
        aU0 = wmma_f16(a00, bu0, aU0);
        aG0 = wmma_f16(a00, bg0, aG0);
        aU1 = wmma_f16(a10, bu0, aU1);
        aG1 = wmma_f16(a10, bg0, aG1);
        aU0 = wmma_f16(a01, bu1, aU0);
        aG0 = wmma_f16(a01, bg1, aG0);
        aU1 = wmma_f16(a11, bu1, aU1);
        aG1 = wmma_f16(a11, bg1, aG1);
        __syncthreads();
    }

    const float bub = (float)ub[(long)e * EW + w0];
    const float bgb = (float)gb[(long)e * EW + w0];
    #pragma unroll
    for (int v = 0; v < 8; ++v) {
        const int m = v + ((lane >> 4) << 3);
        const int r0 = blockIdx.y * 64 + mi * 32 + m;
        const int r1 = r0 + 16;
        const long ta = sparse ? maskr[r0 * T_D + t] : r0;
        const long tb = sparse ? maskr[r1 * T_D + t] : r1;
        const float s0 = rw[ta * NE + e];
        const float s1 = rw[tb * NE + e];
        act[((long)r0 * T_D + t) * EW_EFF + w0] = (f16)(glu_act(aU0[v] + bub, aG0[v] + bgb) * s0);
        act[((long)r1 * T_D + t) * EW_EFF + w0] = (f16)(glu_act(aU1[v] + bub, aG1[v] + bgb) * s1);
    }
}

// ---------------- 5. dense down-projection ----------------
// Block 256: 64 tokens x 64 h. K = 8*2880 (act rows are K-contiguous). A staged in LDS.
__global__ void __launch_bounds__(256, 1)
k_down_dense(const f16* __restrict__ act, const f16* __restrict__ dmat,
             const int* __restrict__ cdcs, float* __restrict__ outacc) {
    __shared__ __align__(16) f16 atile[2][64][BK];
    const int tid = threadIdx.x;
    const int wave = tid >> 5, lane = tid & 31;
    const int mlane = lane & 15;
    const int kbA = (lane >> 4) << 3;
    const int kbB = (lane >> 4) << 4;

    const int srow = tid >> 3;
    const int scol = (tid & 7) * 8;
    const unsigned goff0 = ((unsigned)(blockIdx.y * 64 + srow)      * (T_D * EW_EFF) + scol) * 2u;
    const unsigned goff1 = ((unsigned)(blockIdx.y * 64 + srow + 32) * (T_D * EW_EFF) + scol) * 2u;
    const unsigned long long abase = (unsigned long long)act;
    const unsigned lds0 = (unsigned)(unsigned long long)&atile[0][srow][scol];

    const int ni = wave & 3, mi = wave >> 2;
    const int h0 = blockIdx.x * 64 + ni * 16 + mlane;
    const f16* dbase = dmat + (long)h0 * (NE * EW);
    const int lr0 = mi * 32 + mlane, lr1 = lr0 + 16;

    v8f acc0 = {0,0,0,0,0,0,0,0}, acc1 = acc0;

    async_ld_b128(lds0,        goff0, abase);
    async_ld_b128(lds0 + 4096, goff1, abase);

    const int CPT = EW_EFF / BK;      // 45 chunks per expert
    const int NCH = T_D * CPT;        // 360 total
    for (int t = 0; t < T_D; ++t) {
        const f16* brow = dbase + (long)cdcs[t] * EW;
        for (int j = 0; j < CPT; ++j) {
            const int c = t * CPT + j;
            if (c + 1 < NCH) {
                const unsigned lb = lds0 + (((c + 1) & 1) ? 8192u : 0u);
                const unsigned kB = (unsigned)((c + 1) * BK * 2);
                async_ld_b128(lb,        goff0 + kB, abase);
                async_ld_b128(lb + 4096, goff1 + kB, abase);
                wait_async_le2();
            } else {
                wait_async_0();
            }
            __syncthreads();
            const int kw = j * BK;
            const f16* ap0 = &atile[c & 1][lr0][0];
            const f16* ap1 = &atile[c & 1][lr1][0];
            v16h a00 = a_frag(ap0, 0,  kbA);
            v16h a10 = a_frag(ap1, 0,  kbA);
            v16h a01 = a_frag(ap0, 32, kbA);
            v16h a11 = a_frag(ap1, 32, kbA);
            v16h b0  = *(const v16h*)(brow + kw + 0  + kbB);
            v16h b1  = *(const v16h*)(brow + kw + 32 + kbB);
            acc0 = wmma_f16(a00, b0, acc0);
            acc1 = wmma_f16(a10, b0, acc1);
            acc0 = wmma_f16(a01, b1, acc0);
            acc1 = wmma_f16(a11, b1, acc1);
            __syncthreads();
        }
    }
    #pragma unroll
    for (int v = 0; v < 8; ++v) {
        const int m = v + ((lane >> 4) << 3);
        const int r = blockIdx.y * 64 + mi * 32 + m;
        outacc[(long)r * HID + h0]        = acc0[v];
        outacc[(long)(r + 16) * HID + h0] = acc1[v];
    }
}

// ---------------- 6. sparse down-projection + scatter-add ----------------
__global__ void __launch_bounds__(256, 1)
k_down_sparse(const f16* __restrict__ act, const f16* __restrict__ dmat,
              const int* __restrict__ cdcs, const int* __restrict__ maskr,
              float* __restrict__ outacc) {
    __shared__ __align__(16) f16 atile[2][64][BK];
    const int tid = threadIdx.x;
    const int wave = tid >> 5, lane = tid & 31;
    const int mlane = lane & 15;
    const int kbA = (lane >> 4) << 3;
    const int kbB = (lane >> 4) << 4;
    const int t = blockIdx.z;
    const int e = cdcs[T_D + t];

    const int srow = tid >> 3;
    const int scol = (tid & 7) * 8;
    const unsigned goff0 = (((unsigned)(blockIdx.y * 64 + srow)      * T_D + t) * EW_EFF + scol) * 2u;
    const unsigned goff1 = (((unsigned)(blockIdx.y * 64 + srow + 32) * T_D + t) * EW_EFF + scol) * 2u;
    const unsigned long long abase = (unsigned long long)act;
    const unsigned lds0 = (unsigned)(unsigned long long)&atile[0][srow][scol];

    const int ni = wave & 3, mi = wave >> 2;
    const int h0 = blockIdx.x * 64 + ni * 16 + mlane;
    const f16* brow = dmat + (long)h0 * (NE * EW) + (long)e * EW;
    const int lr0 = mi * 32 + mlane, lr1 = lr0 + 16;

    v8f acc0 = {0,0,0,0,0,0,0,0}, acc1 = acc0;

    async_ld_b128(lds0,        goff0, abase);
    async_ld_b128(lds0 + 4096, goff1, abase);

    const int NCH = EW_EFF / BK;      // 45
    for (int c = 0; c < NCH; ++c) {
        if (c + 1 < NCH) {
            const unsigned lb = lds0 + (((c + 1) & 1) ? 8192u : 0u);
            const unsigned kB = (unsigned)((c + 1) * BK * 2);
            async_ld_b128(lb,        goff0 + kB, abase);
            async_ld_b128(lb + 4096, goff1 + kB, abase);
            wait_async_le2();
        } else {
            wait_async_0();
        }
        __syncthreads();
        const int kc = c * BK;
        const f16* ap0 = &atile[c & 1][lr0][0];
        const f16* ap1 = &atile[c & 1][lr1][0];
        v16h a00 = a_frag(ap0, 0,  kbA);
        v16h a10 = a_frag(ap1, 0,  kbA);
        v16h a01 = a_frag(ap0, 32, kbA);
        v16h a11 = a_frag(ap1, 32, kbA);
        v16h b0  = *(const v16h*)(brow + kc + 0  + kbB);
        v16h b1  = *(const v16h*)(brow + kc + 32 + kbB);
        acc0 = wmma_f16(a00, b0, acc0);
        acc1 = wmma_f16(a10, b0, acc1);
        acc0 = wmma_f16(a01, b1, acc0);
        acc1 = wmma_f16(a11, b1, acc1);
        __syncthreads();
    }
    #pragma unroll
    for (int v = 0; v < 8; ++v) {
        const int m = v + ((lane >> 4) << 3);
        const int r0 = blockIdx.y * 64 + mi * 32 + m;
        const int r1 = r0 + 16;
        const int tokA = maskr[r0 * T_D + t];
        const int tokB = maskr[r1 * T_D + t];
        atomicAdd(&outacc[(long)tokA * HID + h0], acc0[v]);
        atomicAdd(&outacc[(long)tokB * HID + h0], acc1[v]);
    }
}

// ---------------- 7. finalize f32 -> f16 ----------------
__global__ void k_finalize(const float* __restrict__ acc, f16* __restrict__ out, int n) {
    int i = blockIdx.x * blockDim.x + threadIdx.x;
    if (i < n) out[i] = (f16)acc[i];
}

// ---------------- launch ----------------
extern "C" void kernel_launch(void* const* d_in, const int* in_sizes, int n_in,
                              void* d_out, int out_size, void* d_ws, size_t ws_size,
                              hipStream_t stream) {
    (void)in_sizes; (void)n_in; (void)out_size; (void)ws_size;
    const f16* x  = (const f16*)d_in[0];
    const f16* gw = (const f16*)d_in[1];
    const f16* u  = (const f16*)d_in[2];
    const f16* g  = (const f16*)d_in[3];
    const f16* dm = (const f16*)d_in[4];
    const f16* ub = (const f16*)d_in[5];
    const f16* gb = (const f16*)d_in[6];
    f16* out = (f16*)d_out;

    char* ws = (char*)d_ws;
    float* rw     = (float*)(ws + OFF_RW);
    int*   cdcs   = (int*)  (ws + OFF_CDCS);
    int*   maskr  = (int*)  (ws + OFF_MASKR);
    float* outacc = (float*)(ws + OFF_OUTACC);
    f16*   actd   = (f16*)  (ws + OFF_ACTD);
    f16*   acts   = (f16*)  (ws + OFF_ACTS);

    k_router     <<<NTOK, 32, 0, stream>>>(x, gw, rw);
    k_expert_rank<<<1,    32, 0, stream>>>(rw, cdcs);
    k_maskr      <<<T_D, NTOK, 0, stream>>>(rw, cdcs, maskr);

    dim3 gd(45, 16, 8);                       // 2880/64 w-blocks, 1024/64 token-blocks, 8 experts
    k_upgate<<<gd, 256, 0, stream>>>(x, u, g, ub, gb, rw, cdcs, nullptr, actd, 0);
    dim3 gs(45, 4, 8);                        // 256/64 token-blocks
    k_upgate<<<gs, 256, 0, stream>>>(x, u, g, ub, gb, rw, cdcs, maskr,  acts, 1);

    dim3 dd(32, 16, 1);                       // 2048/64 h-blocks, 1024/64 token-blocks
    k_down_dense<<<dd, 256, 0, stream>>>(actd, dm, cdcs, outacc);
    dim3 dsg(32, 4, 8);
    k_down_sparse<<<dsg, 256, 0, stream>>>(acts, dm, cdcs, maskr, outacc);

    k_finalize<<<(NTOK * HID + 255) / 256, 256, 0, stream>>>(outacc, out, NTOK * HID);
}
